// LSTM_91122026152527
// MI455X (gfx1250) — matrix operand order repaired and checked by reading, not compiled, not run on hardware
//
#include <hip/hip_runtime.h>
#include <hip/hip_bf16.h>

// ---------------------------------------------------------------------------
// LSTM forward for MI455X (gfx1250), wave32 + WMMA bf16 + async-LDS staging.
//   B=32, T=2048, C=256, es=256, gates=4*es=1024
// Phase 0: convert x -> bf16, W1/W2 -> transposed bf16 ([N,K]) in workspace
// Phase 1: pre = x @ W1 + b1 + b2  (bf16 WMMA, f32 out, stored as [T,B,1024])
// Phase 2: sequential recurrence, 2 workgroups (one per batch half of 16);
//          W2^T streamed from L2 (double-buffered registers; pinning it in
//          VGPRs was tried and spilled -> reverted), next step's pre tile
//          staged into LDS via global_load_async_to_lds_b128 (ASYNCcnt).
// ---------------------------------------------------------------------------

typedef __attribute__((ext_vector_type(16))) __bf16 v16bf;
typedef __attribute__((ext_vector_type(8)))  __bf16 v8bf;
typedef __attribute__((ext_vector_type(8)))  float  v8f;

#define BB    32
#define TT    2048
#define CC    256
#define ES    256
#define GATES 1024  // 4*ES

// ---- math helpers (VALU transcendental path) ------------------------------
static __device__ __forceinline__ float fast_sigmoid(float x) {
    return 1.0f / (1.0f + __expf(-x));
}
static __device__ __forceinline__ float fast_tanh(float x) {
    x = fminf(15.0f, fmaxf(-15.0f, x));
    float e = __expf(2.0f * x);
    return (e - 1.0f) / (e + 1.0f);
}

// ---- CDNA5 async copy: global -> LDS, 16B per lane, tracked by ASYNCcnt ---
static __device__ __forceinline__ void async_cp16(unsigned lds_off,
                                                  const void* gsrc) {
    asm volatile("global_load_async_to_lds_b128 %0, %1, off"
                 :
                 : "v"(lds_off), "v"((unsigned long long)(size_t)gsrc)
                 : "memory");
}
static __device__ __forceinline__ void wait_async0() {
#if __has_builtin(__builtin_amdgcn_s_wait_asynccnt)
    __builtin_amdgcn_s_wait_asynccnt(0);
#else
    asm volatile("s_wait_asynccnt 0x0" ::: "memory");
#endif
}

// ---- WMMA fragment loaders (16-bit, 16x32 A / 32x16 B lane layouts) -------
// A (row-major [M,K]): lane L holds row (L&15); lanes<16 get K 0..15,
// lanes>=16 get K 16..31 -> two contiguous 16B chunks.
static __device__ __forceinline__ v16bf load_frag_a(const __bf16* p) {
    union { v16bf v; v8bf h[2]; } u;
    u.h[0] = *(const v8bf*)(p);
    u.h[1] = *(const v8bf*)(p + 8);
    return u.v;
}
// B from W^T (row-major [N,K]): lane L holds column (L&15); lanes<16 get
// K k0..k0+15, lanes>=16 get K k0+16..k0+31 -> two contiguous 16B chunks.
static __device__ __forceinline__ v16bf load_frag_b(const __bf16* p) {
    union { v16bf v; v8bf h[2]; } u;
    u.h[0] = *(const v8bf*)(p);
    u.h[1] = *(const v8bf*)(p + 8);
    return u.v;
}

// ---------------------------------------------------------------------------
// Phase 0: dtype conversion + weight transpose
// ---------------------------------------------------------------------------
__global__ void k_convert(const float* __restrict__ x,
                          const float* __restrict__ W1,
                          const float* __restrict__ W2,
                          __bf16* __restrict__ x_bf,   // [B*T, 256]
                          __bf16* __restrict__ W1T,    // [1024, 256]
                          __bf16* __restrict__ W2T)    // [1024, 256]
{
    const long stride = (long)gridDim.x * blockDim.x;
    const long i0 = (long)blockIdx.x * blockDim.x + threadIdx.x;
    const long nx = (long)BB * TT * CC;
    for (long p = i0; p < nx; p += stride)
        x_bf[p] = (__bf16)x[p];
    const long nw = (long)CC * GATES;
    for (long p = i0; p < nw; p += stride) {
        int n = (int)(p >> 8);   // row of transposed layout
        int k = (int)(p & 255);
        W1T[p] = (__bf16)W1[(long)k * GATES + n];
        W2T[p] = (__bf16)W2[(long)k * GATES + n];
    }
}

// ---------------------------------------------------------------------------
// Phase 1: pre = x @ W1 + (b1 + b2).  M=65536, N=1024, K=256.
// Output layout [T, B, 1024] so each timestep's block is contiguous (64 KB
// per batch-half) for async staging in phase 2.
// 8 waves/block; wave owns one 16-row M-tile and 4 N-tiles. Massive TLP
// (65536 waves) hides per-wave load latency here; k-loop double-buffered.
// ---------------------------------------------------------------------------
__global__ void __launch_bounds__(256) k_gemm_pre(
    const __bf16* __restrict__ x_bf,  // [B*T, 256]
    const __bf16* __restrict__ W1T,   // [1024, 256]
    const float* __restrict__ b1,
    const float* __restrict__ b2,
    float* __restrict__ pre)          // [T, B, 1024]
{
    const int lane = threadIdx.x & 31;
    const int wid  = threadIdx.x >> 5;
    const int flat = blockIdx.x * 8 + wid;  // 65536 waves total
    const int mtile = flat >> 4;            // 0..4095
    const int ngrp  = flat & 15;            // 0..15, 64 columns each
    const int col   = lane & 15;

    const int mrow = mtile * 16 + col;
    const __bf16* arow = x_bf + (long)mrow * CC + ((lane & 16) ? 16 : 0);
    const int bk = (lane & 16) ? 16 : 0;

    v8f acc[4];
#pragma unroll
    for (int tn = 0; tn < 4; ++tn) {
        const int n = ngrp * 64 + tn * 16 + col;
        const float bias = b1[n] + b2[n];
#pragma unroll
        for (int r = 0; r < 8; ++r) acc[tn][r] = bias;
    }

    // software-pipelined k-loop
    v16bf a_cur = load_frag_a(arow);
    v16bf b_cur[4];
#pragma unroll
    for (int tn = 0; tn < 4; ++tn) {
        const int n = ngrp * 64 + tn * 16 + col;
        b_cur[tn] = load_frag_b(W1T + (long)n * CC + bk);
    }
#pragma unroll
    for (int k0 = 0; k0 < CC; k0 += 32) {
        v16bf a_nxt = a_cur;
        v16bf b_nxt[4];
#pragma unroll
        for (int tn = 0; tn < 4; ++tn) b_nxt[tn] = b_cur[tn];
        const int k1 = k0 + 32;
        if (k1 < CC) {
            a_nxt = load_frag_a(arow + k1);
#pragma unroll
            for (int tn = 0; tn < 4; ++tn) {
                const int n = ngrp * 64 + tn * 16 + col;
                b_nxt[tn] = load_frag_b(W1T + (long)n * CC + k1 + bk);
            }
        }
#pragma unroll
        for (int tn = 0; tn < 4; ++tn)
            acc[tn] = __builtin_amdgcn_wmma_f32_16x16x32_bf16(
                false, a_cur, false, b_cur[tn], (short)0, acc[tn], false, false);
        a_cur = a_nxt;
#pragma unroll
        for (int tn = 0; tn < 4; ++tn) b_cur[tn] = b_nxt[tn];
    }

    // store to [T, B, 1024]; a 16-row M-tile never crosses a batch boundary
    const int m0    = mtile * 16;
    const int bidx  = m0 >> 11;              // / 2048
    const int tbase = (m0 & 2047) + ((lane >> 4) << 3);
#pragma unroll
    for (int tn = 0; tn < 4; ++tn) {
        const int n = ngrp * 64 + tn * 16 + col;
#pragma unroll
        for (int r = 0; r < 8; ++r)
            pre[((long)(tbase + r) * BB + bidx) * GATES + n] = acc[tn][r];
    }
}

// ---------------------------------------------------------------------------
// Phase 2: sequential LSTM. 2 blocks x 1024 threads (32 waves).
// Block owns batch rows [b0, b0+16). Dynamic LDS (168 KB):
//   [0      , 128K) : pre_lds, double-buffered 16x1024 f32 per step
//   [128K   , 160K) : v_lds, gate preactivations 16x1024 bf16
//   [160K   , 168K) : h_lds, hidden state 16x256 bf16
// Step t: issue async copy of pre(t+1) -> LDS, WMMA-GEMM v = pre_t + h@W2
// (k-loop double-buffered; W2^T streamed from the 192MB L2 — register
// footprint ~110 VGPRs fits the 8-waves/SIMD budget without spilling; a
// fully VGPR-pinned W2 variant spilled and was reverted), gates elementwise
// with c/h state in registers, s_wait_asynccnt + barrier, repeat.
// ---------------------------------------------------------------------------
__global__ void __launch_bounds__(1024) k_lstm_seq(
    const float* __restrict__ pre,    // [T, B, 1024]
    const __bf16* __restrict__ W2T,   // [1024, 256]
    const float* __restrict__ h0,     // [B, 256]
    const float* __restrict__ c0,     // [B, 256]
    float* __restrict__ out1,         // [B, T*256]
    float* __restrict__ out2,         // [T, B, 256]
    float* __restrict__ hout,         // [B, 256]
    float* __restrict__ cout)         // [B, 256]
{
    extern __shared__ char smem[];
    float*  pre_lds = (float*)smem;                  // 2 x 16 x 1024 f32
    __bf16* v_lds   = (__bf16*)(smem + 131072);      // 16 x 1024 bf16
    __bf16* h_lds   = (__bf16*)(smem + 163840);      // 16 x 256 bf16

    const int tid  = threadIdx.x;
    const int lane = tid & 31;
    const int wid  = tid >> 5;        // 0..31
    const int col  = lane & 15;
    const int b0   = blockIdx.x * 16;
    const unsigned pre_lds_base = (unsigned)(size_t)pre_lds;  // LDS byte addr

    // Fixed elementwise ownership: 4 consecutive cells of the 16x256 state.
    float c_reg[4], h_reg[4];
#pragma unroll
    for (int q = 0; q < 4; ++q) {
        const int idx = tid * 4 + q;      // 0..4095
        const int b = idx >> 8, j = idx & 255;
        c_reg[q] = c0[(b0 + b) * ES + j];
        h_reg[q] = h0[(b0 + b) * ES + j];
        h_lds[idx] = (__bf16)h_reg[q];
    }

    // Stage pre(t=0) into buffer 0: 64 KB = 4096 x 16B, 4 chunks per thread.
    {
        const char* g0 = (const char*)(pre + ((long)0 * BB + b0) * GATES);
#pragma unroll
        for (int q = 0; q < 4; ++q) {
            const unsigned byte = (unsigned)((q * 1024 + tid) * 16);
            async_cp16(pre_lds_base + byte, g0 + byte);
        }
    }
    wait_async0();
    __syncthreads();

    const __bf16* arow = h_lds + col * ES + ((lane & 16) ? 16 : 0);
    const int bk = (lane & 16) ? 16 : 0;
    const int rb = (lane >> 4) << 3;

    for (int t = 0; t < TT; ++t) {
        const int buf = t & 1;

        // Kick off async staging of next step's pre tile (disjoint buffer).
        if (t + 1 < TT) {
            const char* gn =
                (const char*)(pre + ((long)(t + 1) * BB + b0) * GATES);
            const unsigned dst = pre_lds_base + (unsigned)((buf ^ 1) * 65536);
#pragma unroll
            for (int q = 0; q < 4; ++q) {
                const unsigned byte = (unsigned)((q * 1024 + tid) * 16);
                async_cp16(dst + byte, gn + byte);
            }
        }

        // ---- GEMM phase: v = pre_t + h @ W2 ----------------------------
        v8f acc[2];
#pragma unroll
        for (int tn = 0; tn < 2; ++tn) {
            const int n = (2 * wid + tn) * 16 + col;
#pragma unroll
            for (int r = 0; r < 8; ++r)
                acc[tn][r] = pre_lds[buf * 16384 + (rb + r) * GATES + n];
        }

        v16bf a_cur = load_frag_a(arow);
        v16bf b_cur[2];
#pragma unroll
        for (int tn = 0; tn < 2; ++tn) {
            const int n = (2 * wid + tn) * 16 + col;
            b_cur[tn] = load_frag_b(W2T + (long)n * ES + bk);
        }
#pragma unroll
        for (int k0 = 0; k0 < ES; k0 += 32) {
            v16bf a_nxt = a_cur;
            v16bf b_nxt[2];
#pragma unroll
            for (int tn = 0; tn < 2; ++tn) b_nxt[tn] = b_cur[tn];
            const int k1 = k0 + 32;
            if (k1 < ES) {
                a_nxt = load_frag_a(arow + k1);
#pragma unroll
                for (int tn = 0; tn < 2; ++tn) {
                    const int n = (2 * wid + tn) * 16 + col;
                    b_nxt[tn] = load_frag_b(W2T + (long)n * ES + k1 + bk);
                }
            }
#pragma unroll
            for (int tn = 0; tn < 2; ++tn)
                acc[tn] = __builtin_amdgcn_wmma_f32_16x16x32_bf16(
                    false, a_cur, false, b_cur[tn], (short)0, acc[tn],
                    false, false);
            a_cur = a_nxt;
#pragma unroll
            for (int tn = 0; tn < 2; ++tn) b_cur[tn] = b_nxt[tn];
        }

#pragma unroll
        for (int tn = 0; tn < 2; ++tn) {
            const int n = (2 * wid + tn) * 16 + col;
#pragma unroll
            for (int r = 0; r < 8; ++r)
                v_lds[(rb + r) * GATES + n] = (__bf16)acc[tn][r];
        }
        __syncthreads();

        // ---- gate elementwise phase ------------------------------------
#pragma unroll
        for (int q = 0; q < 4; ++q) {
            const int idx = tid * 4 + q;
            const int b = idx >> 8, j = idx & 255;
            const float vf = (float)v_lds[b * GATES + j];
            const float vi = (float)v_lds[b * GATES + j + ES];
            const float vo = (float)v_lds[b * GATES + j + 2 * ES];
            const float vg = (float)v_lds[b * GATES + j + 3 * ES];
            const float f  = fast_sigmoid(vf);
            const float ii = fast_sigmoid(vi);
            const float o  = fast_sigmoid(vo);
            const float g  = fast_tanh(vg);
            const float cn = f * c_reg[q] + ii * g;
            const float hn = o * fast_tanh(cn);
            c_reg[q] = cn;
            h_reg[q] = hn;
            h_lds[idx] = (__bf16)hn;
            out2[((long)t * BB + (b0 + b)) * ES + j] = hn;
            out1[(long)(b0 + b) * ((long)TT * ES) + (long)t * ES + j] = hn;
        }
        wait_async0();     // own async copies for t+1 have landed
        __syncthreads();   // all waves' copies landed; h_lds ready
    }

#pragma unroll
    for (int q = 0; q < 4; ++q) {
        const int idx = tid * 4 + q;
        const int b = idx >> 8, j = idx & 255;
        hout[(b0 + b) * ES + j] = h_reg[q];
        cout[(b0 + b) * ES + j] = c_reg[q];
    }
}

// ---------------------------------------------------------------------------
extern "C" void kernel_launch(void* const* d_in, const int* in_sizes, int n_in,
                              void* d_out, int out_size, void* d_ws,
                              size_t ws_size, hipStream_t stream)
{
    (void)in_sizes; (void)n_in; (void)out_size; (void)ws_size;

    const float* x  = (const float*)d_in[0];
    const float* h0 = (const float*)d_in[1];
    const float* c0 = (const float*)d_in[2];
    const float* W1 = (const float*)d_in[3];
    const float* W2 = (const float*)d_in[4];
    const float* b1 = (const float*)d_in[5];
    const float* b2 = (const float*)d_in[6];

    float* out1 = (float*)d_out;                         // [B, T*es]
    float* out2 = out1 + (long)BB * TT * ES;             // [T, B, es]
    float* hout = out2 + (long)TT * BB * ES;             // [B, es]
    float* cout = hout + (long)BB * ES;                  // [B, es]

    char* ws = (char*)d_ws;
    float*  pre  = (float*)ws;                                   // 256 MB
    __bf16* x_bf = (__bf16*)(ws + (size_t)BB * TT * GATES * 4);  //  32 MB
    __bf16* W1T  = x_bf + (size_t)BB * TT * CC;                  // 512 KB
    __bf16* W2T  = W1T + (size_t)CC * GATES;                     // 512 KB

    k_convert<<<2048, 256, 0, stream>>>(x, W1, W2, x_bf, W1T, W2T);
    k_gemm_pre<<<8192, 256, 0, stream>>>(x_bf, W1T, b1, b2, pre);
    k_lstm_seq<<<2, 1024, 172032, stream>>>(pre, W2T, h0, c0,
                                            out1, out2, hout, cout);
}
